// Attention_47132971107029
// MI455X (gfx1250) — matrix operand order repaired and checked by reading, not compiled, not run on hardware
//
#include <hip/hip_runtime.h>
#include <hip/hip_bf16.h>

typedef __bf16 bf16_t;
typedef __attribute__((ext_vector_type(16))) __bf16 v16bf;
typedef __attribute__((ext_vector_type(8)))  __bf16 v8bf;
typedef __attribute__((ext_vector_type(4)))  __bf16 v4bf;
typedef __attribute__((ext_vector_type(8)))  float  v8f;

#define S_   2048
#define B_   2
#define D_   1024
#define H_   16
#define C_   64
#define NROW (S_ * B_)      // 4096
#define SCALE_ 0.125f       // 1/sqrt(64)

// ---------------------------------------------------------------------------
// helpers
// ---------------------------------------------------------------------------
static __device__ __forceinline__ v16bf make_a(v8bf lo, v8bf hi) {
  union { v16bf v; v8bf h[2]; } u;
  u.h[0] = lo; u.h[1] = hi;
  return u.v;
}

static __device__ __forceinline__ v8f wmma_bf16(v16bf a, v16bf b, v8f c) {
  // D = A(16x32 bf16) * B(32x16 bf16) + C(16x16 f32)
  return __builtin_amdgcn_wmma_f32_16x16x32_bf16(false, a, false, b,
                                                 (short)0, c, false, false);
}

// async copy of 16 bytes per lane: LDS[lds_off] = MEM[g .. g+15]  (ASYNCcnt)
static __device__ __forceinline__ void async_cp16(unsigned lds_off,
                                                  const bf16_t* g) {
  asm volatile("global_load_async_to_lds_b128 %0, %1, off"
               :: "v"(lds_off), "v"(g) : "memory");
}
static __device__ __forceinline__ void wait_async0() {
  asm volatile("s_wait_asynccnt 0x0" ::: "memory");
}

// reductions within each 16-lane half of the wave32 (D-matrix row = 16 lanes)
static __device__ __forceinline__ float half_max(float x) {
  x = fmaxf(x, __shfl_xor(x, 1, 32));
  x = fmaxf(x, __shfl_xor(x, 2, 32));
  x = fmaxf(x, __shfl_xor(x, 4, 32));
  x = fmaxf(x, __shfl_xor(x, 8, 32));
  return x;
}
static __device__ __forceinline__ float half_sum(float x) {
  x += __shfl_xor(x, 1, 32);
  x += __shfl_xor(x, 2, 32);
  x += __shfl_xor(x, 4, 32);
  x += __shfl_xor(x, 8, 32);
  return x;
}

// WMMA GEMM mainloop: 16x64 f32 tile, K in steps of 32.
template<int KTOT>
static __device__ __forceinline__ void gemm_acc(const bf16_t* __restrict__ A, int lda,
                                                const bf16_t* __restrict__ Bt, int ldb,
                                                int l15, int hi, v8f acc[4]) {
  const bf16_t* arow = A + (size_t)l15 * lda;
  #pragma unroll 4
  for (int kb = 0; kb < KTOT; kb += 32) {
    // A-operand: lanes0-15 K=0..15, lanes16-31 K=16..31
    v8bf a_lo = *(const v8bf*)(arow + kb + hi * 16);
    v8bf a_hi = *(const v8bf*)(arow + kb + hi * 16 + 8);
    v16bf a = make_a(a_lo, a_hi);
    #pragma unroll
    for (int j = 0; j < 4; ++j) {
      // B-operand: lanes0-15 hold K=0..15 (n=lane), lanes16-31 K=16..31
      v16bf b = *(const v16bf*)(Bt + (size_t)(j * 16 + l15) * ldb + kb + hi * 16);
      acc[j] = wmma_bf16(a, b, acc[j]);
    }
  }
}

// ---------------------------------------------------------------------------
// conversion kernels
// ---------------------------------------------------------------------------
__global__ __launch_bounds__(256) void cvt4_kernel(const float* __restrict__ src,
                                                   bf16_t* __restrict__ dst, int n4) {
  int i = blockIdx.x * blockDim.x + threadIdx.x;
  if (i < n4) {
    float4 f = ((const float4*)src)[i];
    v4bf o = { (bf16_t)f.x, (bf16_t)f.y, (bf16_t)f.z, (bf16_t)f.w };
    ((v4bf*)dst)[i] = o;
  }
}

// 1024x1024 f32 -> bf16 transpose (LDS-tiled, both sides coalesced)
__global__ __launch_bounds__(256) void transpose_cvt_kernel(const float* __restrict__ w,
                                                            bf16_t* __restrict__ wt) {
  __shared__ bf16_t tile[32][33];
  const int bx = blockIdx.x * 32, by = blockIdx.y * 32;
  const int tx = threadIdx.x, ty = threadIdx.y;        // block (32, 8)
  #pragma unroll
  for (int i = 0; i < 32; i += 8)
    tile[ty + i][tx] = (bf16_t)w[(size_t)(by + ty + i) * D_ + bx + tx];
  __syncthreads();
  #pragma unroll
  for (int i = 0; i < 32; i += 8)
    wt[(size_t)(bx + ty + i) * D_ + by + tx] = tile[tx][ty + i];
}

// ---------------------------------------------------------------------------
// QKV projection + RMSNorm + RoPE (blockIdx.z: 0=Q, 1=K, 2=V)
// ---------------------------------------------------------------------------
__global__ __launch_bounds__(128) void qkv_kernel(
    const bf16_t* __restrict__ xb,
    const bf16_t* __restrict__ wqt, const bf16_t* __restrict__ wkt,
    const bf16_t* __restrict__ wvt,
    const float* __restrict__ qnw, const float* __restrict__ knw,
    const float* __restrict__ rope,
    bf16_t* __restrict__ qout, bf16_t* __restrict__ kout,
    bf16_t* __restrict__ vtout)
{
  const int lane = threadIdx.x & 31;
  const int wid  = threadIdx.x >> 5;
  const int l15  = lane & 15, hi = lane >> 4;
  const int mode = blockIdx.z;
  const int h    = blockIdx.x;                 // head == 64-wide N tile
  const int m0   = (blockIdx.y * 4 + wid) * 16;

  const bf16_t* wt = (mode == 0) ? wqt : (mode == 1) ? wkt : wvt;

  v8f acc[4];
  #pragma unroll
  for (int j = 0; j < 4; ++j) { v8f z = {}; acc[j] = z; }

  gemm_acc<D_>(xb + (size_t)m0 * D_, D_, wt + (size_t)h * C_ * D_, D_, l15, hi, acc);

  if (mode == 2) {
    // V: store transposed bf16 [b,h,c,s] for the PV B-operand
    #pragma unroll
    for (int r = 0; r < 8; ++r) {
      const int m = r + 8 * hi;
      const int rid = m0 + m;
      const int s = rid / B_, b = rid % B_;
      #pragma unroll
      for (int j = 0; j < 4; ++j) {
        const int c = j * 16 + l15;
        vtout[((size_t)(b * H_ + h) * C_ + c) * S_ + s] = (bf16_t)acc[j][r];
      }
    }
  } else {
    const float* nw = (mode == 0) ? qnw : knw;
    bf16_t* outp = (mode == 0) ? qout : kout;
    float w[4];
    #pragma unroll
    for (int j = 0; j < 4; ++j) w[j] = nw[j * 16 + l15];
    #pragma unroll
    for (int r = 0; r < 8; ++r) {
      float ss = 0.f;
      #pragma unroll
      for (int j = 0; j < 4; ++j) ss += acc[j][r] * acc[j][r];
      ss = half_sum(ss);                       // sum of squares over the 64 channels
      const float rms = rsqrtf(ss * (1.0f / (float)C_) + 1e-6f);
      const int m = r + 8 * hi;
      const int rid = m0 + m;
      const int s = rid / B_, b = rid % B_;
      float vals[4];
      #pragma unroll
      for (int j = 0; j < 4; ++j) vals[j] = acc[j][r] * rms * w[j];
      bf16_t* dst = outp + ((size_t)(b * H_ + h) * S_ + s) * C_;
      #pragma unroll
      for (int jj = 0; jj < 2; ++jj) {         // RoPE: pair (c, c+32)
        const int i = jj * 16 + l15;
        const float* rp = rope + ((size_t)s * (C_ / 2) + i) * 4;
        const float cs = rp[0];                // [s,i,0,0] = cos
        const float sn = rp[2];                // [s,i,1,0] = sin
        const float t0 = vals[jj], t1 = vals[jj + 2];
        dst[i]          = (bf16_t)(cs * t0 - sn * t1);
        dst[i + C_ / 2] = (bf16_t)(sn * t0 + cs * t1);
      }
    }
  }
}

// ---------------------------------------------------------------------------
// Flash attention: 4 waves share (b,h); K/V tiles staged in LDS via the
// async-to-LDS engine (ASYNCcnt), double-buffered, one barrier per key block.
// ---------------------------------------------------------------------------
__global__ __launch_bounds__(128) void attn_kernel(
    const bf16_t* __restrict__ qb, const bf16_t* __restrict__ kb,
    const bf16_t* __restrict__ vt, bf16_t* __restrict__ ao)
{
  __shared__ bf16_t ktile[2][32 * 64];         // 32 keys x 64 ch, row-major
  __shared__ bf16_t vtile[2][64 * 32];         // 64 ch x 32 keys, row-major
  __shared__ bf16_t pbuf[4][16 * 32];          // per-wave P staging tile

  const int tid  = threadIdx.x;
  const int lane = tid & 31;
  const int wid  = tid >> 5;
  const int l15  = lane & 15, hi = lane >> 4;
  const int bh   = blockIdx.y;                 // b*H + h
  const int b    = bh / H_, h = bh % H_;
  const int s0   = (blockIdx.x * 4 + wid) * 16;

  const bf16_t* kbase = kb + (size_t)bh * S_ * C_;
  const bf16_t* vbase = vt + (size_t)bh * C_ * S_;

  // stage one K/V block (32 keys) into LDS buffer `buf` with async copies
  auto stage = [&](int buf, int t0) {
    // K tile: 32 rows x 64 ch = 4KB, fully contiguous in global
    const bf16_t* kg = kbase + (size_t)t0 * C_;
    #pragma unroll
    for (int i = 0; i < 2; ++i) {
      int idx = (i * 128 + tid) * 8;           // 8 bf16 = 16B per lane
      async_cp16((unsigned)(size_t)(&ktile[buf][idx]), kg + idx);
    }
    // V^T tile: 64 rows (c) x 32 t; global row stride S_, LDS row stride 32
    const bf16_t* vg = vbase + t0;
    #pragma unroll
    for (int i = 0; i < 2; ++i) {
      int idx = i * 128 + tid;                 // 0..255 16B chunks
      int row = idx >> 2, col = (idx & 3) * 8;
      async_cp16((unsigned)(size_t)(&vtile[buf][row * 32 + col]),
                 vg + (size_t)row * S_ + col);
    }
  };

  // Q tile as two A-operands (K = c 0..31, 32..63)
  const bf16_t* qrow = qb + ((size_t)bh * S_ + s0) * C_ + (size_t)l15 * C_;
  v16bf qa0 = make_a(*(const v8bf*)(qrow + hi * 16),
                     *(const v8bf*)(qrow + hi * 16 + 8));
  v16bf qa1 = make_a(*(const v8bf*)(qrow + 32 + hi * 16),
                     *(const v8bf*)(qrow + 40 + hi * 16));

  v8f acc[4];
  #pragma unroll
  for (int j = 0; j < 4; ++j) { v8f z = {}; acc[j] = z; }
  float mi[8], li[8];
  #pragma unroll
  for (int r = 0; r < 8; ++r) { mi[r] = -1e30f; li[r] = 0.f; }

  stage(0, 0);                                 // prologue

  for (int t0 = 0, it = 0; t0 < S_; t0 += 32, ++it) {
    const int cur = it & 1;
    wait_async0();                             // own async loads for `cur` done
    __syncthreads();                           // all waves' loads visible; prior
                                               // readers of `cur^1` are past it
    if (t0 + 32 < S_) stage(cur ^ 1, t0 + 32); // prefetch next block

    const bf16_t* kt  = &ktile[cur][0];
    const bf16_t* vtl = &vtile[cur][0];

    // K B-operands from LDS: two 16-key tiles x two 32-channel K chunks
    v16bf kb00 = *(const v16bf*)(kt + l15 * 64 + hi * 16);
    v16bf kb01 = *(const v16bf*)(kt + l15 * 64 + 32 + hi * 16);
    v16bf kb10 = *(const v16bf*)(kt + (16 + l15) * 64 + hi * 16);
    v16bf kb11 = *(const v16bf*)(kt + (16 + l15) * 64 + 32 + hi * 16);

    v8f sc0 = {}; sc0 = wmma_bf16(qa0, kb00, sc0); sc0 = wmma_bf16(qa1, kb01, sc0);
    v8f sc1 = {}; sc1 = wmma_bf16(qa0, kb10, sc1); sc1 = wmma_bf16(qa1, kb11, sc1);

    // online softmax + stage P into LDS (D-layout -> row-major bf16)
    #pragma unroll
    for (int r = 0; r < 8; ++r) {
      const float a0 = sc0[r] * SCALE_;
      const float a1 = sc1[r] * SCALE_;
      float mx = half_max(fmaxf(a0, a1));
      const float mn = fmaxf(mi[r], mx);
      const float corr = __expf(mi[r] - mn);
      const float p0 = __expf(a0 - mn);
      const float p1 = __expf(a1 - mn);
      const float rs = half_sum(p0 + p1);
      li[r] = li[r] * corr + rs;
      mi[r] = mn;
      #pragma unroll
      for (int j = 0; j < 4; ++j) acc[j][r] *= corr;
      const int m = r + 8 * hi;
      pbuf[wid][m * 32 + l15]      = (bf16_t)p0;
      pbuf[wid][m * 32 + 16 + l15] = (bf16_t)p1;
    }
    asm volatile("" ::: "memory");             // keep ds_loads after ds_stores

    // reload P as A-operand (16x32, K = 32 keys)
    v16bf pA = make_a(*(const v8bf*)(&pbuf[wid][l15 * 32 + hi * 16]),
                      *(const v8bf*)(&pbuf[wid][l15 * 32 + hi * 16 + 8]));

    // PV: 4 channel chunks of 16, B-operand from LDS V^T (contiguous in t)
    #pragma unroll
    for (int j = 0; j < 4; ++j) {
      v16bf vb = *(const v16bf*)(vtl + (j * 16 + l15) * 32 + hi * 16);
      acc[j] = wmma_bf16(pA, vb, acc[j]);
    }
  }

  // normalize and write attention output bf16 [s,b,h*C] for the out-proj GEMM
  #pragma unroll
  for (int r = 0; r < 8; ++r) {
    const int m = r + 8 * hi;
    const int s = s0 + m;
    const float inv = 1.0f / li[r];
    #pragma unroll
    for (int j = 0; j < 4; ++j) {
      ao[((size_t)s * B_ + b) * D_ + h * C_ + j * 16 + l15] =
          (bf16_t)(acc[j][r] * inv);
    }
  }
}

// ---------------------------------------------------------------------------
// output projection: (4096 x 1024) @ (1024 x 1024) -> f32
// ---------------------------------------------------------------------------
__global__ __launch_bounds__(128) void proj_kernel(
    const bf16_t* __restrict__ ao, const bf16_t* __restrict__ wot,
    float* __restrict__ out)
{
  const int lane = threadIdx.x & 31;
  const int wid  = threadIdx.x >> 5;
  const int l15  = lane & 15, hi = lane >> 4;
  const int n0   = blockIdx.x * 64;
  const int m0   = (blockIdx.y * 4 + wid) * 16;

  v8f acc[4];
  #pragma unroll
  for (int j = 0; j < 4; ++j) { v8f z = {}; acc[j] = z; }

  gemm_acc<D_>(ao + (size_t)m0 * D_, D_, wot + (size_t)n0 * D_, D_, l15, hi, acc);

  #pragma unroll
  for (int r = 0; r < 8; ++r) {
    const int rid = m0 + r + 8 * hi;
    #pragma unroll
    for (int j = 0; j < 4; ++j)
      out[(size_t)rid * D_ + n0 + j * 16 + l15] = acc[j][r];
  }
}

// ---------------------------------------------------------------------------
// host launch
// ---------------------------------------------------------------------------
extern "C" void kernel_launch(void* const* d_in, const int* in_sizes, int n_in,
                              void* d_out, int out_size, void* d_ws, size_t ws_size,
                              hipStream_t stream) {
  (void)in_sizes; (void)n_in; (void)out_size; (void)ws_size;
  const float* x    = (const float*)d_in[0];
  const float* rope = (const float*)d_in[1];
  const float* Wq   = (const float*)d_in[2];
  const float* Wk   = (const float*)d_in[3];
  const float* Wv   = (const float*)d_in[4];
  const float* qnw  = (const float*)d_in[5];
  const float* knw  = (const float*)d_in[6];
  const float* Wout = (const float*)d_in[7];
  float* out = (float*)d_out;

  // workspace partition (bf16 elements)
  bf16_t* ws = (bf16_t*)d_ws;
  const size_t M1 = (size_t)D_ * D_;           // 1M
  bf16_t* xb  = ws;                            // 4096*1024
  bf16_t* wqt = xb  + (size_t)NROW * D_;
  bf16_t* wkt = wqt + M1;
  bf16_t* wvt = wkt + M1;
  bf16_t* wot = wvt + M1;
  bf16_t* qb  = wot + M1;                      // B*H*S*C = 4M
  bf16_t* kbuf= qb  + (size_t)B_ * H_ * S_ * C_;
  bf16_t* vt  = kbuf+ (size_t)B_ * H_ * S_ * C_;
  bf16_t* ao  = vt  + (size_t)B_ * H_ * S_ * C_;

  // 1) convert x to bf16
  {
    int n4 = (NROW * D_) / 4;
    cvt4_kernel<<<(n4 + 255) / 256, 256, 0, stream>>>(x, xb, n4);
  }
  // 2) transpose+convert weights
  {
    dim3 g(D_ / 32, D_ / 32), blk(32, 8);
    transpose_cvt_kernel<<<g, blk, 0, stream>>>(Wq,   wqt);
    transpose_cvt_kernel<<<g, blk, 0, stream>>>(Wk,   wkt);
    transpose_cvt_kernel<<<g, blk, 0, stream>>>(Wv,   wvt);
    transpose_cvt_kernel<<<g, blk, 0, stream>>>(Wout, wot);
  }
  // 3) QKV projections + RMSNorm + RoPE
  qkv_kernel<<<dim3(H_, NROW / 64, 3), 128, 0, stream>>>(
      xb, wqt, wkt, wvt, qnw, knw, rope, qb, kbuf, vt);
  // 4) flash attention (async-LDS staged K/V)
  attn_kernel<<<dim3(S_ / 64, B_ * H_), 128, 0, stream>>>(qb, kbuf, vt, ao);
  // 5) output projection
  proj_kernel<<<dim3(D_ / 64, NROW / 64), 128, 0, stream>>>(ao, wot, out);
}
